// ChebConvConvolutional_66554813219093
// MI455X (gfx1250) — matrix operand order repaired and checked by reading, not compiled, not run on hardware
//
#include <hip/hip_runtime.h>
#include <hip/hip_bf16.h>

typedef __bf16 bf16_t;
typedef __attribute__((ext_vector_type(16))) __bf16 v16bf;
typedef __attribute__((ext_vector_type(8)))  float  v8f;

#define NNODES 10000
#define NEDGES 160000
#define FDIM   512
#define DOUT   256

// ---------------------------------------------------------------------------
// A-fragment builder: 16-bit A 16x32 WMMA layout. Lane holds one 16-element
// contiguous K run of its row: [k0+half*16, +16),
// loaded as fp32 (4x float4) and packed to bf16 (v_cvt_pk_bf16_f32).
// ---------------------------------------------------------------------------
__device__ __forceinline__ v16bf a_frag(const float* __restrict__ arow,
                                        int k0, int aoff) {
  float4 a0 = *(const float4*)(arow + k0 + aoff);
  float4 a1 = *(const float4*)(arow + k0 + aoff + 4);
  float4 a2 = *(const float4*)(arow + k0 + aoff + 8);
  float4 a3 = *(const float4*)(arow + k0 + aoff + 12);
  v16bf v;
  v[0]=(bf16_t)a0.x;  v[1]=(bf16_t)a0.y;  v[2]=(bf16_t)a0.z;  v[3]=(bf16_t)a0.w;
  v[4]=(bf16_t)a1.x;  v[5]=(bf16_t)a1.y;  v[6]=(bf16_t)a1.z;  v[7]=(bf16_t)a1.w;
  v[8]=(bf16_t)a2.x;  v[9]=(bf16_t)a2.y;  v[10]=(bf16_t)a2.z; v[11]=(bf16_t)a2.w;
  v[12]=(bf16_t)a3.x; v[13]=(bf16_t)a3.y; v[14]=(bf16_t)a3.z; v[15]=(bf16_t)a3.w;
  return v;
}

// ---------------------------------------------------------------------------
// GEMM: D[M,N] = A[M,K] (fp32 row-major) @ B (stored BT[N,K] bf16 row-major)
//       (+ Cin if non-null).
// Register-blocked: each wave owns a 32x64 output tile = 2x4 grid of 16x16
// WMMA accumulators; 8 independent v_wmma_f32_16x16x32_bf16 per K-step.
// Block = 8 waves as 4(M) x 2(N) -> 128x128 block tile.
// K % 32 == 0, N % 128 == 0; M tail handled by clamp+store-guard.
// ---------------------------------------------------------------------------
__global__ __launch_bounds__(256)
void gemm_wmma_bf16(const float* __restrict__ A, const bf16_t* __restrict__ BT,
                    const float* __restrict__ Cin, float* __restrict__ D,
                    int M, int K, int N) {
  const int lane = threadIdx.x & 31;
  const int wave = threadIdx.x >> 5;
  const int wrow = wave >> 1;            // 0..3 (M)
  const int wcol = wave & 1;             // 0..1 (N)
  const int half = lane >> 4;            // 0: lanes 0-15, 1: lanes 16-31
  const int l15  = lane & 15;

  const int m0 = (blockIdx.y * 8 + wrow * 2) * 16;  // wave rows [m0, m0+32)
  const int n0 = (blockIdx.x * 2 + wcol) * 64;      // wave cols [n0, n0+64)
  if (m0 >= M) return;                   // wave-uniform: EXEC stays all-1s

  // A row pointers (clamped for the M tail; WMMA still executes, stores guard)
  int r0 = m0 + l15;       if (r0 >= M) r0 = M - 1;
  int r1 = m0 + 16 + l15;  if (r1 >= M) r1 = M - 1;
  const float* arow0 = A + (size_t)r0 * K;
  const float* arow1 = A + (size_t)r1 * K;
  const int aoff = half * 16;

  // B fragment pointers: lane = column n0 + j*16 + l15, 16 contiguous bf16
  // K values (32B) starting at half*16.
  const bf16_t* brow0 = BT + (size_t)(n0 +  0 + l15) * K + half * 16;
  const bf16_t* brow1 = BT + (size_t)(n0 + 16 + l15) * K + half * 16;
  const bf16_t* brow2 = BT + (size_t)(n0 + 32 + l15) * K + half * 16;
  const bf16_t* brow3 = BT + (size_t)(n0 + 48 + l15) * K + half * 16;

  v8f acc[2][4];
#pragma unroll
  for (int t = 0; t < 2; ++t)
#pragma unroll
    for (int j = 0; j < 4; ++j)
      acc[t][j] = (v8f){0.f, 0.f, 0.f, 0.f, 0.f, 0.f, 0.f, 0.f};

  if (Cin) {
#pragma unroll
    for (int t = 0; t < 2; ++t)
#pragma unroll
      for (int i = 0; i < 8; ++i) {
        int row = m0 + t * 16 + half * 8 + i;
        if (row < M) {
          const float* cp = Cin + (size_t)row * N + n0 + l15;
#pragma unroll
          for (int j = 0; j < 4; ++j) acc[t][j][i] = cp[j * 16];
        }
      }
  }

  for (int k0 = 0; k0 < K; k0 += 32) {
    v16bf af0 = a_frag(arow0, k0, aoff);
    v16bf af1 = a_frag(arow1, k0, aoff);
    v16bf bf0 = *(const v16bf*)(brow0 + k0);
    v16bf bf1 = *(const v16bf*)(brow1 + k0);
    v16bf bf2 = *(const v16bf*)(brow2 + k0);
    v16bf bf3 = *(const v16bf*)(brow3 + k0);

    if (k0 + 32 < K) {
      __builtin_prefetch(arow0 + k0 + 32 + aoff, 0, 1);
      __builtin_prefetch(arow1 + k0 + 32 + aoff, 0, 1);
    }

    // 8 independent WMMAs -> back-to-back XDL issue, no D->A/B hazards
    acc[0][0] = __builtin_amdgcn_wmma_f32_16x16x32_bf16(false, af0, false, bf0, (short)0, acc[0][0], false, false);
    acc[0][1] = __builtin_amdgcn_wmma_f32_16x16x32_bf16(false, af0, false, bf1, (short)0, acc[0][1], false, false);
    acc[0][2] = __builtin_amdgcn_wmma_f32_16x16x32_bf16(false, af0, false, bf2, (short)0, acc[0][2], false, false);
    acc[0][3] = __builtin_amdgcn_wmma_f32_16x16x32_bf16(false, af0, false, bf3, (short)0, acc[0][3], false, false);
    acc[1][0] = __builtin_amdgcn_wmma_f32_16x16x32_bf16(false, af1, false, bf0, (short)0, acc[1][0], false, false);
    acc[1][1] = __builtin_amdgcn_wmma_f32_16x16x32_bf16(false, af1, false, bf1, (short)0, acc[1][1], false, false);
    acc[1][2] = __builtin_amdgcn_wmma_f32_16x16x32_bf16(false, af1, false, bf2, (short)0, acc[1][2], false, false);
    acc[1][3] = __builtin_amdgcn_wmma_f32_16x16x32_bf16(false, af1, false, bf3, (short)0, acc[1][3], false, false);
  }

#pragma unroll
  for (int t = 0; t < 2; ++t)
#pragma unroll
    for (int i = 0; i < 8; ++i) {
      int row = m0 + t * 16 + half * 8 + i;
      if (row < M) {
        float* dp = D + (size_t)row * N + n0 + l15;
#pragma unroll
        for (int j = 0; j < 4; ++j) dp[j * 16] = acc[t][j][i];
      }
    }
}

// ---------------------------------------------------------------------------
// Weight prep: W[K,N] fp32 row-major  ->  WT[N,K] bf16 row-major
// ---------------------------------------------------------------------------
__global__ void wconv_kernel(const float* __restrict__ W, bf16_t* __restrict__ WT,
                             int K, int N) {
  int idx = blockIdx.x * blockDim.x + threadIdx.x;
  if (idx >= K * N) return;
  int k = idx / N, n = idx % N;
  WT[(size_t)n * K + k] = (bf16_t)W[idx];
}

// ---------------------------------------------------------------------------
// Degree / norm precompute (edge_index is int64 per the reference)
// ---------------------------------------------------------------------------
__global__ void init_deg_kernel(float* deg_g, float* deg_c, int n) {
  int i = blockIdx.x * blockDim.x + threadIdx.x;
  if (i < n) { deg_g[i] = 1.0f; deg_c[i] = 0.0f; }   // GCN self-loop weight 1
}

__global__ void deg_accum_kernel(const long long* __restrict__ ei,
                                 const float* __restrict__ w,
                                 float* deg_g, float* deg_c, int E) {
  int e = blockIdx.x * blockDim.x + threadIdx.x;
  if (e >= E) return;
  int r = (int)ei[e], c = (int)ei[E + e];
  float ww = w[e];
  atomicAdd(deg_g + c, ww);                 // GCN: segment by col
  if (r != c) atomicAdd(deg_c + r, ww);     // Cheb: self-loops removed, by row
}

__global__ void dinv_kernel(float* deg_g, float* deg_c, int n) {
  int i = blockIdx.x * blockDim.x + threadIdx.x;
  if (i >= n) return;
  float dg = deg_g[i], dc = deg_c[i];
  deg_g[i] = dg > 0.f ? rsqrtf(dg) : 0.f;   // overwrite in place with dinv
  deg_c[i] = dc > 0.f ? rsqrtf(dc) : 0.f;
}

__global__ void norm_kernel(const long long* __restrict__ ei,
                            const float* __restrict__ w,
                            const float* __restrict__ dinv_g,
                            const float* __restrict__ dinv_c,
                            float* norm_g, float* norm_c, int E) {
  int e = blockIdx.x * blockDim.x + threadIdx.x;
  if (e >= E) return;
  int r = (int)ei[e], c = (int)ei[E + e];
  float wv = w[e];
  norm_g[e] = dinv_g[r] * wv * dinv_g[c];
  float w0 = (r == c) ? 0.f : wv;
  norm_c[e] = dinv_c[r] * w0 * dinv_c[c];
}

// ---------------------------------------------------------------------------
// Aggregation: one block per edge; coalesced float4 gather of the feature row
// + global_atomic_add_f32 scatter. dst[col] += coef*norm[e]*src[row].
// ---------------------------------------------------------------------------
__global__ __launch_bounds__(128)
void scatter_kernel(const long long* __restrict__ ei,
                    const float* __restrict__ norm,
                    const float* __restrict__ src, float* __restrict__ dst,
                    int E, int F, float coef) {
  int e = blockIdx.x;
  float s = coef * norm[e];
  if (s == 0.f) return;                      // block-uniform skip (self loops)
  int r = (int)ei[e], c = (int)ei[E + e];
  const float4* sp = (const float4*)(src + (size_t)r * F);
  float* dp = dst + (size_t)c * F;
  for (int q = threadIdx.x; q < (F >> 2); q += blockDim.x) {
    float4 v = sp[q];
    atomicAdd(dp + q * 4 + 0, s * v.x);
    atomicAdd(dp + q * 4 + 1, s * v.y);
    atomicAdd(dp + q * 4 + 2, s * v.z);
    atomicAdd(dp + q * 4 + 3, s * v.w);
  }
}

// GCN aggregation init with the self-loop term: dst = dinv^2 * h
__global__ void selfloop_init_kernel(const float* __restrict__ h,
                                     const float* __restrict__ dinv,
                                     float* __restrict__ dst, int n, int F) {
  int idx = blockIdx.x * blockDim.x + threadIdx.x;
  if (idx >= n * F) return;
  float d = dinv[idx / F];
  dst[idx] = d * d * h[idx];
}

__global__ void fill_zero_kernel(float* p, int n) {
  int i = blockIdx.x * blockDim.x + threadIdx.x;
  if (i < n) p[i] = 0.f;
}

__global__ void neg_copy_kernel(const float* __restrict__ s, float* __restrict__ d, int n) {
  int i = blockIdx.x * blockDim.x + threadIdx.x;
  if (i < n) d[i] = -s[i];
}

// out = celu(agg + bias)   (alpha = 1)
__global__ void bias_celu_kernel(const float* __restrict__ agg,
                                 const float* __restrict__ b,
                                 float* __restrict__ out, int n, int F) {
  int idx = blockIdx.x * blockDim.x + threadIdx.x;
  if (idx >= n * F) return;
  float z = agg[idx] + b[idx % F];
  out[idx] = z > 0.f ? z : (expf(z) - 1.f);
}

// ---------------------------------------------------------------------------
extern "C" void kernel_launch(void* const* d_in, const int* in_sizes, int n_in,
                              void* d_out, int out_size, void* d_ws, size_t ws_size,
                              hipStream_t stream) {
  const float*     x   = (const float*)d_in[0];
  const long long* ei  = (const long long*)d_in[1];   // int64 in reference
  const float*     ew  = (const float*)d_in[2];
  const float*     W1  = (const float*)d_in[3];
  const float*     b1  = (const float*)d_in[4];
  const float*     W2  = (const float*)d_in[5];
  const float*     b2  = (const float*)d_in[6];
  const float*     Wc  = (const float*)d_in[7];       // [3, 512, 256]
  const float*     bc  = (const float*)d_in[8];
  float*           out = (float*)d_out;

  const int n = NNODES, E = NEDGES, F = FDIM, DO = DOUT;
  const size_t nf = (size_t)n * F;        // node-feature buffer (fp32 elems)
  const size_t no = (size_t)n * DO;

  // --- workspace carving (fp32 first, then bf16 weights) ---
  float* bufA   = (float*)d_ws;           // nf
  float* bufB   = bufA + nf;              // nf
  float* bufC   = bufB + nf;              // nf
  float* outc   = bufC + nf;              // no
  float* dinv_g = outc + no;              // n   (deg -> dinv in place)
  float* dinv_c = dinv_g + n;             // n
  float* norm_g = dinv_c + n;             // E
  float* norm_c = norm_g + E;             // E
  bf16_t* w1t   = (bf16_t*)(norm_c + E);  // 512*512
  bf16_t* w2t   = w1t + (size_t)F * F;    // 512*512
  bf16_t* wct   = w2t + (size_t)F * F;    // 3 * 256*512

  const int TB = 256;
  const int gN  = (int)((nf + TB - 1) / TB);
  const int gNo = (int)((no + TB - 1) / TB);
  const int gE  = (E + TB - 1) / TB;
  const int gn  = (n + TB - 1) / TB;
  const dim3 gemm_block(256);
  const dim3 gemm512(F / 128, (n + 127) / 128);   // 4 x 79
  const dim3 gemm256(DO / 128, (n + 127) / 128);  // 2 x 79

  // 1) degrees + symmetric norms
  init_deg_kernel<<<gn, TB, 0, stream>>>(dinv_g, dinv_c, n);
  deg_accum_kernel<<<gE, TB, 0, stream>>>(ei, ew, dinv_g, dinv_c, E);
  dinv_kernel<<<gn, TB, 0, stream>>>(dinv_g, dinv_c, n);
  norm_kernel<<<gE, TB, 0, stream>>>(ei, ew, dinv_g, dinv_c, norm_g, norm_c, E);

  // 2) weight convert + transpose to bf16 [N,K]
  wconv_kernel<<<(F * F + TB - 1) / TB, TB, 0, stream>>>(W1, w1t, F, F);
  wconv_kernel<<<(F * F + TB - 1) / TB, TB, 0, stream>>>(W2, w2t, F, F);
  for (int k = 0; k < 3; ++k)
    wconv_kernel<<<(F * DO + TB - 1) / TB, TB, 0, stream>>>(
        Wc + (size_t)k * F * DO, wct + (size_t)k * DO * F, F, DO);

  // 3) GCN layer 1: lin -> sym-norm aggregate -> bias+celu
  gemm_wmma_bf16<<<gemm512, gemm_block, 0, stream>>>(x, w1t, nullptr, bufA, n, F, F);
  selfloop_init_kernel<<<gN, TB, 0, stream>>>(bufA, dinv_g, bufB, n, F);
  scatter_kernel<<<E, 128, 0, stream>>>(ei, norm_g, bufA, bufB, E, F, 1.0f);
  bias_celu_kernel<<<gN, TB, 0, stream>>>(bufB, b1, bufA, n, F);        // h1 = bufA

  // 4) GCN layer 2
  gemm_wmma_bf16<<<gemm512, gemm_block, 0, stream>>>(bufA, w2t, nullptr, bufB, n, F, F);
  selfloop_init_kernel<<<gN, TB, 0, stream>>>(bufB, dinv_g, bufC, n, F);
  scatter_kernel<<<E, 128, 0, stream>>>(ei, norm_g, bufB, bufC, E, F, 1.0f);
  bias_celu_kernel<<<gN, TB, 0, stream>>>(bufC, b2, bufB, n, F);        // h2 = bufB

  // 5) ChebConv (sym, lambda_max=2): L_hat = -D^-1/2 A D^-1/2
  //    Tx0 = h2, Tx1 = lhat(h2), Tx2 = 2*lhat(Tx1) - Tx0
  fill_zero_kernel<<<gN, TB, 0, stream>>>(bufA, (int)nf);               // Tx1 = bufA
  scatter_kernel<<<E, 128, 0, stream>>>(ei, norm_c, bufB, bufA, E, F, -1.0f);
  neg_copy_kernel<<<gN, TB, 0, stream>>>(bufB, bufC, (int)nf);          // Tx2 = bufC
  scatter_kernel<<<E, 128, 0, stream>>>(ei, norm_c, bufA, bufC, E, F, -2.0f);

  gemm_wmma_bf16<<<gemm256, gemm_block, 0, stream>>>(bufB, wct,                  nullptr, outc, n, F, DO);
  gemm_wmma_bf16<<<gemm256, gemm_block, 0, stream>>>(bufA, wct + (size_t)DO * F, outc,    outc, n, F, DO);
  gemm_wmma_bf16<<<gemm256, gemm_block, 0, stream>>>(bufC, wct + (size_t)2 * DO * F, outc, outc, n, F, DO);

  bias_celu_kernel<<<gNo, TB, 0, stream>>>(outc, bc, out, n, DO);
}
